// SoftmaxMatcher_8675833938308
// MI455X (gfx1250) — matrix operand order repaired and checked by reading, not compiled, not run on hardware
//
#include <hip/hip_runtime.h>

#define DEVFN __device__ __forceinline__

typedef unsigned int       u32;
typedef unsigned long long u64;
typedef unsigned short     u16;
typedef __attribute__((ext_vector_type(4)))  u32    v4u;
typedef __attribute__((ext_vector_type(4)))  int    v4i;
typedef __attribute__((ext_vector_type(8)))  int    v8i;
typedef __attribute__((ext_vector_type(8)))  float  v8f;
typedef __attribute__((ext_vector_type(16))) __bf16 v16bf;

struct V8x32 { v4u lo, hi; };                 // 32 bytes, feeds one v16bf operand
static_assert(sizeof(V8x32) == 32, "V8x32 size");
static_assert(sizeof(v16bf) == 32, "v16bf size");

#if __has_builtin(__builtin_amdgcn_tensor_load_to_lds)
#define USE_TDM 1
#else
#define USE_TDM 0
#endif

constexpr int kBN = 4;        // window pairs (B=8, ws=2)
constexpr int kN  = 1024;     // keypoints per image
constexpr int kC  = 256;      // descriptor channels
constexpr int kH  = 128, kW = 128;
constexpr int kM  = kH * kW;  // 16384 dense positions
constexpr float kTempScale = 144.2695040888963f;   // (1/0.01) * log2(e)

// ---- workspace layout (bytes) ----
constexpr size_t WS_SRCW   = 0;                                   // bf16 [4*1024][256]
constexpr size_t WS_TGTW   = WS_SRCW + (size_t)kBN * kN * kC * 2; // bf16 [4*16384][256]
constexpr size_t WS_INVSRC = WS_TGTW + (size_t)kBN * kM * kC * 2; // f32  [4*1024]
constexpr size_t WS_INVTGT = WS_INVSRC + (size_t)kBN * kN * 4;    // f32  [4*16384]

DEVFN u16 f2bf(float x) {
  u32 u = __builtin_bit_cast(u32, x);
  u = (u + 0x7FFFu + ((u >> 16) & 1u)) >> 16;   // round-to-nearest-even
  return (u16)u;
}

// ---------------------------------------------------------------------------
// LDS tile geometry: 64 rows (m) x 512B (256 bf16) per stage.
//  - TDM path: TDM pads 4 dwords every 128 dwords -> 528B row stride, which
//    makes the per-lane strided B reads bank-conflict-free with no swizzle.
//  - Fallback: 512B stride + XOR swizzle of 16B chunks.
// ---------------------------------------------------------------------------
#if USE_TDM
constexpr int kRowStride = 528;
DEVFN int ldsOff(int row, int c16) { return row * kRowStride + (c16 << 4); }
#else
constexpr int kRowStride = 512;
DEVFN int ldsOff(int row, int c16) { return row * kRowStride + ((c16 ^ (row & 15)) << 4); }
#endif

#if USE_TDM
// Issue one TDM load of a 64-row x 512B tile (contiguous in global) into LDS
// at byte offset ldsOffBytes, with 16B padding appended to each 512B row.
DEVFN void tdm_load_tile(const void* gsrc, u32 ldsOffBytes) {
  u64 ga = (u64)(size_t)gsrc;
  v4u g0;
  g0[0] = 1u;                                            // count=1, user desc
  g0[1] = ldsOffBytes;                                   // lds_addr
  g0[2] = (u32)ga;                                       // global_addr[31:0]
  g0[3] = (u32)((ga >> 32) & 0x01FFFFFFu) | (2u << 30);  // addr[56:32] | type=2
  v8i g1;
  // d0: workgroup_mask=0 | data_size=4B(2) | pad_enable | pad_interval=128dw(6)
  //     | pad_amount=4dw(3)
  g1[0] = (int)((2u << 16) | (1u << 20) | (6u << 22) | (3u << 25));
  const u32 td0 = 128u;                                  // tensor_dim0 (dwords/row)
  const u32 td1 = 1u << 20;                              // tensor_dim1 (rows, ample)
  g1[1] = (int)((td0 & 0xFFFFu) << 16);                  // abar_addr=0 | td0.lo
  g1[2] = (int)(((td0 >> 16) & 0xFFFFu) | ((td1 & 0xFFFFu) << 16));
  g1[3] = (int)(((td1 >> 16) & 0xFFFFu) | (128u << 16)); // td1.hi | tile_dim0=128
  g1[4] = (int)64u;                                      // tile_dim1=64, tile_dim2=0
  g1[5] = (int)128u;                                     // tensor_dim0_stride=128 dw
  g1[6] = 0;                                             // stride0.hi | stride1.lo
  g1[7] = 0;                                             // stride1.hi
  v4i z4 = {0, 0, 0, 0};
#if defined(__clang_major__) && (__clang_major__ >= 23)
  v8i z8 = {0, 0, 0, 0, 0, 0, 0, 0};
  __builtin_amdgcn_tensor_load_to_lds(g0, g1, z4, z4, z8, 0);
#else
  __builtin_amdgcn_tensor_load_to_lds(g0, g1, z4, z4, 0);
#endif
}
DEVFN void tdm_wait() {
#if __has_builtin(__builtin_amdgcn_s_wait_tensorcnt)
  __builtin_amdgcn_s_wait_tensorcnt(0);
#else
  asm volatile("s_wait_tensorcnt 0x0" ::: "memory");
#endif
}
#endif

// ---------------------------------------------------------------------------
// K1: normalize src keypoint descriptors -> bf16 [point][k], store inv norms
// ---------------------------------------------------------------------------
__global__ __launch_bounds__(256) void sm_prep_src(
    const float* __restrict__ kdesc, u16* __restrict__ srcW, float* __restrict__ invSrc)
{
  int idx = blockIdx.x * 256 + threadIdx.x;     // 0..4095
  int b = idx >> 10, n = idx & (kN - 1);
  const float* base = kdesc + (size_t)(2 * b) * kC * kN + n;
  float s = 0.f;
  for (int c = 0; c < kC; ++c) { float v = base[(size_t)c * kN]; s += v * v; }
  float inv = rsqrtf(fmaxf(s, 1e-24f));
  invSrc[idx] = inv;
  u16* w = srcW + (size_t)idx * kC;
  for (int c8 = 0; c8 < kC / 8; ++c8) {
    v4u pk;
#pragma unroll
    for (int q = 0; q < 4; ++q) {
      float a  = base[(size_t)(c8 * 8 + 2 * q)     * kN] * inv;
      float bb = base[(size_t)(c8 * 8 + 2 * q + 1) * kN] * inv;
      pk[q] = (u32)f2bf(a) | ((u32)f2bf(bb) << 16);
    }
    *reinterpret_cast<v4u*>(w + c8 * 8) = pk;
  }
}

// ---------------------------------------------------------------------------
// K2: single pass over tgt dense descriptors: raw bf16 [point][k] + inv norms
// ---------------------------------------------------------------------------
__global__ __launch_bounds__(256) void sm_prep_tgt(
    const float* __restrict__ ddense, u16* __restrict__ tgtW, float* __restrict__ invTgt)
{
  int idx = blockIdx.x * 256 + threadIdx.x;     // 0..65535
  int b = idx >> 14, m = idx & (kM - 1);
  const float* base = ddense + (size_t)(2 * b + 1) * kC * kM + m;
  u16* w = tgtW + (size_t)idx * kC;
  float s = 0.f;
  for (int c8 = 0; c8 < kC / 8; ++c8) {
    v4u pk;
#pragma unroll
    for (int q = 0; q < 4; ++q) {
      float a  = base[(size_t)(c8 * 8 + 2 * q)     * kM];
      float bb = base[(size_t)(c8 * 8 + 2 * q + 1) * kM];
      s += a * a + bb * bb;
      pk[q] = (u32)f2bf(a) | ((u32)f2bf(bb) << 16);
    }
    *reinterpret_cast<v4u*>(w + c8 * 8) = pk;
  }
  invTgt[idx] = rsqrtf(fmaxf(s, 1e-24f));
}

// ---------------------------------------------------------------------------
// K3: fused bf16-WMMA GEMM + online softmax + weighted coordinate expectation
//   block = 128 threads (4 waves). Each wave owns 16 keypoint rows; the block
//   streams all 16384 tgt columns in 64-column LDS stages, double buffered,
//   staged by the Tensor Data Mover when available.
// ---------------------------------------------------------------------------
__global__ __launch_bounds__(128) void sm_match_softmax(
    const u16* __restrict__ srcW, const u16* __restrict__ tgtW,
    const float* __restrict__ invTgt, float* __restrict__ outCoords)
{
  constexpr int SUB      = 4;                   // 16-col subtiles per stage
  constexpr int STAGE_M  = SUB * 16;            // 64 columns per stage
  constexpr int NSTAGE   = kM / STAGE_M;        // 256
  constexpr int STAGE_B  = STAGE_M * kRowStride;

  __shared__ __align__(16) char lds[2 * STAGE_B];

  // Escape the LDS array's address so a clobbering asm is a potential writer;
  // without this, AA proves nothing can store to `lds` (the TDM write is an
  // invisible side effect) and all LDS reads fold to undef, deleting the
  // B-feed and most of the WMMAs.
  void* ldsEsc = (void*)lds;
#define LDS_PUBLISH() asm volatile("" :: "r"(ldsEsc) : "memory")

  const int tid    = threadIdx.x;
  const int lane   = tid & 31;
  const int laneLo = lane & 15;
  const int laneHi = lane >> 4;
  const int wave   = tid >> 5;

  const int blk  = blockIdx.x;                  // 64 blocks
  const int b    = blk >> 4;                    // batch pair
  const int n0   = (blk & 15) * 64 + wave * 16; // this wave's 16 rows

  // ---- load A tile (16 rows x 256 K, bf16) into registers, WMMA A layout ----
  v16bf Atile[8];
  const u16* aBase = srcW + (size_t)(b * kN + n0 + laneLo) * kC;
#pragma unroll
  for (int kc = 0; kc < 8; ++kc) {
    V8x32 s;
    s.lo = *reinterpret_cast<const v4u*>(aBase + kc * 32 + laneHi * 8);
    s.hi = *reinterpret_cast<const v4u*>(aBase + kc * 32 + 16 + laneHi * 8);
    Atile[kc] = __builtin_bit_cast(v16bf, s);
  }

  float run_max[8], run_sum[8], run_cx[8], run_cy[8];
#pragma unroll
  for (int r = 0; r < 8; ++r) { run_max[r] = -3.0e38f; run_sum[r] = 0.f; run_cx[r] = 0.f; run_cy[r] = 0.f; }

  const u16* gTile = tgtW + (size_t)b * kM * kC;   // [m][k] contiguous

  // ---- prologue: stage tile 0 into buffer 0 ----
#if USE_TDM
  if (tid == 0) {
    // Visible dummy stores into the (never-read) TDM pad bytes: independently
    // blocks the "never-stored global -> undef" transform.
    *reinterpret_cast<u32*>(lds + 512) = 0u;
    *reinterpret_cast<u32*>(lds + STAGE_B + 512) = 0u;
  }
  if (tid < 32) {                                // wave 0 drives the TDM
    tdm_load_tile(gTile, 0u);
    tdm_wait();
  }
#else
  {
    const v4u* g = reinterpret_cast<const v4u*>(gTile);
#pragma unroll
    for (int i = 0; i < 16; ++i) {
      int q = tid + i * 128;                     // 16B-chunk index, 0..2047
      *reinterpret_cast<v4u*>(lds + ldsOff(q >> 5, q & 31)) = g[q];
    }
  }
#endif
  __syncthreads();
  LDS_PUBLISH();                                 // DMA-written LDS is now "real"

  for (int t = 0; t < NSTAGE; ++t) {
    const int  buf = t & 1;
    char* const lb = lds + buf * STAGE_B;

    // ---- stage next tile into the other buffer (async on TDM path) ----
#if USE_TDM
    if (tid < 32 && t + 1 < NSTAGE)
      tdm_load_tile(gTile + (size_t)(t + 1) * STAGE_M * kC, (u32)((buf ^ 1) * STAGE_B));
#else
    if (t + 1 < NSTAGE) {
      const v4u* g = reinterpret_cast<const v4u*>(gTile + (size_t)(t + 1) * STAGE_M * kC);
      char* const nb = lds + (buf ^ 1) * STAGE_B;
#pragma unroll
      for (int i = 0; i < 16; ++i) {
        int q = tid + i * 128;
        *reinterpret_cast<v4u*>(nb + ldsOff(q >> 5, q & 31)) = g[q];
      }
      if (t + 2 < NSTAGE)
        __builtin_prefetch(gTile + (size_t)(t + 2) * STAGE_M * kC + (size_t)tid * 128, 0, 0);
    }
#endif

    // per-column scale: inverse tgt norm folded with softmax temp (log2 domain)
    float scl[SUB];
#pragma unroll
    for (int s = 0; s < SUB; ++s)
      scl[s] = invTgt[(size_t)b * kM + t * STAGE_M + s * 16 + laneLo] * kTempScale;

    // ---- 32 WMMAs: 4 column subtiles x 8 K-chunks, B fragments double
    //      buffered so ds_load latency hides under the WMMA chain ----
    v8f acc[SUB];
#pragma unroll
    for (int s = 0; s < SUB; ++s) {
      const int row = s * 16 + laneLo;           // B column = lane, row-major K
      V8x32 bf0, bf1;
      bf0.lo = *reinterpret_cast<const v4u*>(lb + ldsOff(row, laneHi));
      bf0.hi = *reinterpret_cast<const v4u*>(lb + ldsOff(row, laneHi + 2));
      v8f c = {0.f, 0.f, 0.f, 0.f, 0.f, 0.f, 0.f, 0.f};
#pragma unroll
      for (int kc = 0; kc < 8; ++kc) {
        if (kc + 1 < 8) {
          bf1.lo = *reinterpret_cast<const v4u*>(lb + ldsOff(row, (kc + 1) * 4 + laneHi));
          bf1.hi = *reinterpret_cast<const v4u*>(lb + ldsOff(row, (kc + 1) * 4 + laneHi + 2));
        }
        v16bf B = __builtin_bit_cast(v16bf, bf0);
        c = __builtin_amdgcn_wmma_f32_16x16x32_bf16(false, Atile[kc], false, B,
                                                    (short)0, c, false, false);
        bf0 = bf1;
      }
      acc[s] = c;
    }

    // ---- online softmax update (C row r lives on 16 lanes of one half-wave) ----
    const int mb = t * STAGE_M + laneLo;
#pragma unroll
    for (int r = 0; r < 8; ++r) {
      float v0 = acc[0][r] * scl[0];
      float v1 = acc[1][r] * scl[1];
      float v2 = acc[2][r] * scl[2];
      float v3 = acc[3][r] * scl[3];
      float tmax = fmaxf(fmaxf(v0, v1), fmaxf(v2, v3));
#pragma unroll
      for (int ms = 1; ms <= 8; ms <<= 1) tmax = fmaxf(tmax, __shfl_xor(tmax, ms, 32));
      float nmax = fmaxf(run_max[r], tmax);
      float sc = exp2f(run_max[r] - nmax);
      float p0 = exp2f(v0 - nmax), p1 = exp2f(v1 - nmax);
      float p2 = exp2f(v2 - nmax), p3 = exp2f(v3 - nmax);
      float ps  = p0 + p1 + p2 + p3;
      float pcx = 0.f, pcy = 0.f;
      {
        int m0 = mb;      pcx += p0 * (float)(m0 & 127); pcy += p0 * (float)(m0 >> 7);
        int m1 = mb + 16; pcx += p1 * (float)(m1 & 127); pcy += p1 * (float)(m1 >> 7);
        int m2 = mb + 32; pcx += p2 * (float)(m2 & 127); pcy += p2 * (float)(m2 >> 7);
        int m3 = mb + 48; pcx += p3 * (float)(m3 & 127); pcy += p3 * (float)(m3 >> 7);
      }
#pragma unroll
      for (int ms = 1; ms <= 8; ms <<= 1) {
        ps  += __shfl_xor(ps,  ms, 32);
        pcx += __shfl_xor(pcx, ms, 32);
        pcy += __shfl_xor(pcy, ms, 32);
      }
      run_sum[r] = run_sum[r] * sc + ps;
      run_cx[r]  = run_cx[r]  * sc + pcx;
      run_cy[r]  = run_cy[r]  * sc + pcy;
      run_max[r] = nmax;
    }

#if USE_TDM
    if (tid < 32 && t + 1 < NSTAGE) tdm_wait();  // next stage resident in LDS
#endif
    __syncthreads();
    LDS_PUBLISH();                               // republish freshly staged buffer
  }
#undef LDS_PUBLISH

  if (laneLo == 0) {
#pragma unroll
    for (int r = 0; r < 8; ++r) {
      int n = n0 + r + 8 * laneHi;               // C layout: VGPR r -> M=r / r+8
      float inv = 1.0f / run_sum[r];
      float* o = outCoords + ((size_t)b * kN + n) * 2;
      o[0] = run_cx[r] * inv;                    // expected x (col)
      o[1] = run_cy[r] * inv;                    // expected y (row)
    }
  }
}

// ---------------------------------------------------------------------------
// K4: bilinear sampling (zeros padding, align_corners=False) + weight fusion.
//   One wave per keypoint; 8 channels per lane.
// ---------------------------------------------------------------------------
__global__ __launch_bounds__(256) void sm_sample_final(
    const float* __restrict__ kscores, const float* __restrict__ kdesc,
    const float* __restrict__ sdense,  const float* __restrict__ ddense,
    const float* __restrict__ invSrc,  const float* __restrict__ coords,
    float* __restrict__ outW)
{
  int lane = threadIdx.x & 31;
  int p = blockIdx.x * 8 + (threadIdx.x >> 5);   // 0..4095
  int b = p >> 10, n = p & (kN - 1);

  float px = coords[(size_t)p * 2 + 0];
  float py = coords[(size_t)p * 2 + 1];
  // grid_sample coordinate mapping: x = ((u_norm+1)*W - 1)/2, u_norm = 2*px/127 - 1
  float fx = px * (128.f / 127.f) - 0.5f;
  float fy = py * (128.f / 127.f) - 0.5f;
  float x0f = floorf(fx), y0f = floorf(fy);
  int   x0 = (int)x0f,   y0 = (int)y0f;
  float wx1 = fx - x0f, wx0 = 1.f - wx1;
  float wy1 = fy - y0f, wy0 = 1.f - wy1;
  int   xi[2] = {x0, x0 + 1}, yi[2] = {y0, y0 + 1};
  float wx[2] = {wx0, wx1},   wy[2] = {wy0, wy1};
  int xc[2], yc[2];
#pragma unroll
  for (int i = 0; i < 2; ++i) {
    xc[i] = min(max(xi[i], 0), kW - 1);
    yc[i] = min(max(yi[i], 0), kH - 1);
  }
  float cw[4];
  int k = 0;
#pragma unroll
  for (int iy = 0; iy < 2; ++iy)
#pragma unroll
    for (int ix = 0; ix < 2; ++ix) {
      bool inb = (xi[ix] >= 0) && (xi[ix] < kW) && (yi[iy] >= 0) && (yi[iy] < kH);
      cw[k++] = wx[ix] * wy[iy] * (inb ? 1.f : 0.f);
    }
  int o00 = yc[0] * kW + xc[0], o01 = yc[0] * kW + xc[1];
  int o10 = yc[1] * kW + xc[0], o11 = yc[1] * kW + xc[1];

  const float* tgt = ddense + (size_t)(2 * b + 1) * kC * kM;   // raw tgt descs
  const float* src = kdesc  + (size_t)(2 * b) * kC * kN + n;
  float invn = invSrc[p];
  float dot = 0.f;
#pragma unroll
  for (int ci = 0; ci < kC / 32; ++ci) {
    int c = lane + ci * 32;
    const float* tc = tgt + (size_t)c * kM;
    float pd = cw[0] * tc[o00] + cw[1] * tc[o01] + cw[2] * tc[o10] + cw[3] * tc[o11];
    dot += pd * (src[(size_t)c * kN] * invn);
  }
#pragma unroll
  for (int ms = 1; ms < 32; ms <<= 1) dot += __shfl_xor(dot, ms, 32);

  if (lane == 0) {
    const float* sc2 = sdense + (size_t)(2 * b + 1) * kM;
    float pscore = cw[0] * sc2[o00] + cw[1] * sc2[o01] + cw[2] * sc2[o10] + cw[3] * sc2[o11];
    float sscore = kscores[(size_t)(2 * b) * kN + n];
    outW[(size_t)b * kN + n] = 0.5f * (dot * (1.f / kC) + 1.f) * sscore * pscore;
  }
}

// ---------------------------------------------------------------------------
extern "C" void kernel_launch(void* const* d_in, const int* in_sizes, int n_in,
                              void* d_out, int out_size, void* d_ws, size_t ws_size,
                              hipStream_t stream) {
  (void)in_sizes; (void)n_in; (void)out_size; (void)ws_size;
  const float* kscores = (const float*)d_in[0];   // (8,1,1024)
  const float* kdesc   = (const float*)d_in[1];   // (8,256,1024)
  const float* sdense  = (const float*)d_in[2];   // (8,1,128,128)
  const float* ddense  = (const float*)d_in[3];   // (8,256,128,128)

  char* ws = (char*)d_ws;
  u16*   srcW   = (u16*)(ws + WS_SRCW);
  u16*   tgtW   = (u16*)(ws + WS_TGTW);
  float* invSrc = (float*)(ws + WS_INVSRC);
  float* invTgt = (float*)(ws + WS_INVTGT);

  float* outCoords = (float*)d_out;                       // (4,1024,2)
  float* outW      = (float*)d_out + (size_t)kBN * kN * 2;// (4,1,1024)

  sm_prep_src<<<dim3(16),  dim3(256), 0, stream>>>(kdesc, srcW, invSrc);
  sm_prep_tgt<<<dim3(256), dim3(256), 0, stream>>>(ddense, tgtW, invTgt);
  sm_match_softmax<<<dim3(64), dim3(128), 0, stream>>>(srcW, tgtW, invTgt, outCoords);
  sm_sample_final<<<dim3(512), dim3(256), 0, stream>>>(kscores, kdesc, sdense, ddense,
                                                       invSrc, outCoords, outW);
}
